// GAT_76184129896720
// MI455X (gfx1250) — matrix operand order repaired and checked by reading, not compiled, not run on hardware
//
#include <hip/hip_runtime.h>
#include <hip/hip_bf16.h>

#define NN      30000
#define EE      480000
#define EMB     256
#define GG      512
#define LL      3
#define AVOCAB  128
#define BVOCAB  16
#define NEGSLOPE 0.2f
// key of -inf under the order-preserving float->uint map
#define NEGINF_KEY 0x007FFFFFu

typedef __attribute__((ext_vector_type(16))) __bf16 v16bf;
typedef __attribute__((ext_vector_type(8)))  float  v8f;

__device__ __forceinline__ unsigned fkey(float f) {
    unsigned u = __float_as_uint(f);
    return (u & 0x80000000u) ? ~u : (u | 0x80000000u);
}
__device__ __forceinline__ float kinv(unsigned k) {
    return __uint_as_float((k & 0x80000000u) ? (k & 0x7FFFFFFFu) : ~k);
}

// ---------------- Atom encoder: h[n][c] = sum_i atom_emb[i][x[n,i]][c] --------
__global__ __launch_bounds__(256) void k_atom(const int* __restrict__ x,
                                              const float* __restrict__ emb,
                                              float* __restrict__ h) {
    __shared__ int xi[9];
    int n = blockIdx.x, c = threadIdx.x;
    if (c < 9) xi[c] = x[n * 9 + c];
    __syncthreads();
    float s = 0.f;
#pragma unroll
    for (int i = 0; i < 9; ++i)
        s += emb[((size_t)i * AVOCAB + xi[i]) * EMB + c];
    h[(size_t)n * EMB + c] = s;
}

// ---------------- Per-layer scratch init ------------------------------------
__global__ __launch_bounds__(256) void k_init_layer(float* __restrict__ outb,
                                                    float* __restrict__ deg,
                                                    float* __restrict__ sums,
                                                    float* __restrict__ denom,
                                                    unsigned* __restrict__ mkey) {
    int idx = blockIdx.x * 256 + threadIdx.x;
    outb[idx] = 0.f;                 // N*EMB elements exactly
    if (idx < NN) {
        deg[idx] = 0.f; sums[idx] = 0.f; denom[idx] = 0.f;
        mkey[idx] = NEGINF_KEY;
    }
}

// -------- wa = We_l @ a_e (3-vec);  lut[j*16+v] = bond_emb[l][j][v] . wa -----
__global__ void k_wa(const float* __restrict__ We_l, const float* __restrict__ ae,
                     const float* __restrict__ bond_l, float* __restrict__ lut) {
    __shared__ float wa[3];
    int t = threadIdx.x;
    if (t < 3) {
        float s = 0.f;
        for (int c = 0; c < EMB; ++c) s += We_l[t * EMB + c] * ae[c];
        wa[t] = s;
    }
    __syncthreads();
    if (t < 48) {
        int j = t >> 4, v = t & 15;
        const float* b = bond_l + (j * BVOCAB + v) * 3;
        lut[t] = b[0] * wa[0] + b[1] * wa[1] + b[2] * wa[2];
    }
}

// -------- per-edge bond score, degree, segment-sum of scores by dst ---------
__global__ __launch_bounds__(256) void k_edge_score(const int* __restrict__ ei,
                                                    const int* __restrict__ attr,
                                                    const float* __restrict__ lut,
                                                    float* __restrict__ escore,
                                                    float* __restrict__ deg,
                                                    float* __restrict__ sums) {
    int i = blockIdx.x * 256 + threadIdx.x;
    if (i >= EE) return;
    float es = lut[attr[i * 3 + 0]] + lut[16 + attr[i * 3 + 1]] + lut[32 + attr[i * 3 + 2]];
    escore[i] = es;
    int d = ei[EE + i];
    atomicAdd(deg + d, 1.f);
    atomicAdd(sums + d, es);
}

// ---------------- WMMA GEMM: C[M,256] = A[M,256] @ B[256,256] (+bias) -------
// 8 waves/block. Tile M=16 x N=128: A tile staged once per K-step into LDS and
// reused by all 8 waves; B staged transposed [n][k] so each lane's fragment is
// one contiguous 32B run (-> ds_load_b128 on both WMMA operands).
template <bool HAS_BIAS>
__global__ __launch_bounds__(256) void k_gemm_wmma(const float* __restrict__ A,
                                                   const float* __restrict__ B,
                                                   const float* __restrict__ bias,
                                                   float* __restrict__ C) {
    __shared__ __bf16 As[16 * 32];    // [m][k]
    __shared__ __bf16 Bt[128 * 32];   // [n][k]  (transposed)
    const int tid = threadIdx.x, lane = tid & 31, wave = tid >> 5;
    const int tm = blockIdx.x;            // M/16 tiles
    const int n0 = blockIdx.y * 128;      // N block base (256/128 = 2)
    const int mrow = lane & 15, kg = lane >> 4;
    v8f acc = {};
    for (int k0 = 0; k0 < EMB; k0 += 32) {
        // stage A 16x32: 512 elems, 2 per thread (coalesced float2 rows)
        {
            int idx = tid * 2;
            int r = idx >> 5, c = idx & 31;
            const float* ap = A + (size_t)(tm * 16 + r) * EMB + k0 + c;
            As[idx]     = (__bf16)ap[0];
            As[idx + 1] = (__bf16)ap[1];
        }
        // stage B 32x128 transposed: 8 threads per k-row, 16 n-values each
        {
            int k  = tid >> 3;
            int nn = (tid & 7) * 16;
            const float* bp = B + (size_t)(k0 + k) * EMB + n0 + nn;
#pragma unroll
            for (int i = 0; i < 16; ++i)
                Bt[(nn + i) * 32 + k] = (__bf16)bp[i];
        }
        __syncthreads();
        // fragments in contiguous k order, k = 16*(lane/16) + i, same for A and B (A 16x32 / B 32x16, wave32)
        v16bf a, b;
#pragma unroll
        for (int i = 0; i < 16; ++i) {
            int ka = (i < 8) ? (kg * 8 + i) : (16 + kg * 8 + (i - 8));
            a[i] = As[mrow * 32 + kg * 16 + i];
            b[i] = Bt[(wave * 16 + mrow) * 32 + kg * 16 + i];
        }
        acc = __builtin_amdgcn_wmma_f32_16x16x32_bf16(false, a, false, b,
                                                      (short)0, acc, false, false);
        __syncthreads();
    }
    // D layout: VGPR r -> M = r + 8*(lane/16), N = lane%16 (per wave's N slice)
    const int col = n0 + wave * 16 + mrow;
    const float bv = HAS_BIAS ? bias[col] : 0.f;
#pragma unroll
    for (int r = 0; r < 8; ++r)
        C[(size_t)(tm * 16 + kg * 8 + r) * EMB + col] = acc[r] + bv;
}

// -------- per-node attention dots: asrc/adst = xl . att_{src,dst} -----------
__global__ __launch_bounds__(256) void k_dots(const float* __restrict__ xl,
                                              const float* __restrict__ as_,
                                              const float* __restrict__ ad_,
                                              float* __restrict__ asrc,
                                              float* __restrict__ adst) {
    int node = blockIdx.x * 8 + (threadIdx.x >> 5);
    int lane = threadIdx.x & 31;
    if (node >= NN) return;
    float s1 = 0.f, s2 = 0.f;
    const float* row = xl + (size_t)node * EMB;
    for (int c = lane; c < EMB; c += 32) {
        float v = row[c];
        s1 += v * as_[c];
        s2 += v * ad_[c];
    }
#pragma unroll
    for (int off = 16; off; off >>= 1) {
        s1 += __shfl_down(s1, off);
        s2 += __shfl_down(s2, off);
    }
    if (lane == 0) { asrc[node] = s1; adst[node] = s2; }
}

// -------- pass 1: alpha + leaky_relu, segment max (ordered-uint atomicMax) --
__global__ __launch_bounds__(256) void k_alpha_max(const int* __restrict__ ei,
                                                   const float* __restrict__ escore,
                                                   const float* __restrict__ deg,
                                                   const float* __restrict__ sums,
                                                   const float* __restrict__ asrc,
                                                   const float* __restrict__ adst,
                                                   float* __restrict__ alpha,
                                                   unsigned* __restrict__ mkey) {
    int i = blockIdx.x * 256 + threadIdx.x;
    if (i >= EE + NN) return;
    int d; float a;
    if (i < EE) {
        int s = ei[i]; d = ei[EE + i];
        a = asrc[s] + adst[d] + escore[i];
    } else {
        int n = i - EE; d = n;
        float ls = sums[n] / fmaxf(deg[n], 1.f);   // self-loop attr = mean of incoming
        a = asrc[n] + adst[n] + ls;
    }
    a = (a >= 0.f) ? a : NEGSLOPE * a;
    alpha[i] = a;
    atomicMax(mkey + d, fkey(a));
}

// -------- pass 2: p = exp(alpha - max); segment sum of p --------------------
__global__ __launch_bounds__(256) void k_exp_sum(const int* __restrict__ ei,
                                                 float* __restrict__ alpha,
                                                 const unsigned* __restrict__ mkey,
                                                 float* __restrict__ denom) {
    int i = blockIdx.x * 256 + threadIdx.x;
    if (i >= EE + NN) return;
    int d = (i < EE) ? ei[EE + i] : (i - EE);
    float p = expf(alpha[i] - kinv(mkey[d]));
    alpha[i] = p;
    atomicAdd(denom + d, p);
}

// -------- pass 3: out[dst] += (p/denom[dst]) * xl[src] ----------------------
__global__ __launch_bounds__(64) void k_scatter(const int* __restrict__ ei,
                                                const float* __restrict__ alpha,
                                                const float* __restrict__ denom,
                                                const float* __restrict__ xl,
                                                float* __restrict__ outb) {
    int i = blockIdx.x;
    int s, d;
    if (i < EE) { s = ei[i]; d = ei[EE + i]; }
    else        { s = d = i - EE; }
    float w = alpha[i] / denom[d];
    const float4* xr = (const float4*)(xl + (size_t)s * EMB);
    float4 v = xr[threadIdx.x];
    float* o = outb + (size_t)d * EMB + threadIdx.x * 4;
    atomicAdd(o + 0, w * v.x);
    atomicAdd(o + 1, w * v.y);
    atomicAdd(o + 2, w * v.z);
    atomicAdd(o + 3, w * v.w);
}

// -------- bias + (relu) + residual, in place into h -------------------------
__global__ __launch_bounds__(256) void k_post(const float* __restrict__ outb,
                                              const float* __restrict__ bias,
                                              float* __restrict__ h, int do_relu) {
    int idx = blockIdx.x * 256 + threadIdx.x;   // N*EMB exactly
    int c = idx & (EMB - 1);
    float v = outb[idx] + bias[c];
    if (do_relu) v = fmaxf(v, 0.f);
    h[idx] = v + h[idx];
}

// ---------------- pooling ----------------------------------------------------
__global__ __launch_bounds__(256) void k_pool_init(float* __restrict__ hg,
                                                   float* __restrict__ cnt) {
    int idx = blockIdx.x * 256 + threadIdx.x;   // G*EMB exactly
    hg[idx] = 0.f;
    if (idx < GG) cnt[idx] = 0.f;
}
__global__ __launch_bounds__(256) void k_pool_cnt(const int* __restrict__ batch,
                                                  float* __restrict__ cnt) {
    int n = blockIdx.x * 256 + threadIdx.x;
    if (n < NN) atomicAdd(cnt + batch[n], 1.f);
}
__global__ __launch_bounds__(256) void k_pool_sum(const int* __restrict__ batch,
                                                  const float* __restrict__ h,
                                                  float* __restrict__ hg) {
    int idx = blockIdx.x * 256 + threadIdx.x;   // N*EMB exactly
    int n = idx >> 8, c = idx & (EMB - 1);
    atomicAdd(hg + (size_t)batch[n] * EMB + c, h[idx]);
}
__global__ __launch_bounds__(256) void k_pool_fin(float* __restrict__ hg,
                                                  const float* __restrict__ cnt) {
    int idx = blockIdx.x * 256 + threadIdx.x;   // G*EMB exactly
    hg[idx] /= fmaxf(cnt[idx >> 8], 1.f);
}

// -------- head tail: out[g] = t[g] . W2 + b2 --------------------------------
__global__ __launch_bounds__(256) void k_head(const float* __restrict__ t,
                                              const float* __restrict__ W2,
                                              const float* __restrict__ b2,
                                              float* __restrict__ out) {
    __shared__ float red[256];
    int g = blockIdx.x, c = threadIdx.x;
    red[c] = t[(size_t)g * EMB + c] * W2[c];
    __syncthreads();
    for (int s = 128; s; s >>= 1) {
        if (c < s) red[c] += red[c + s];
        __syncthreads();
    }
    if (c == 0) out[g] = red[0] + b2[0];
}

extern "C" void kernel_launch(void* const* d_in, const int* in_sizes, int n_in,
                              void* d_out, int out_size, void* d_ws, size_t ws_size,
                              hipStream_t stream) {
    (void)in_sizes; (void)n_in; (void)out_size; (void)ws_size;
    const int*   x        = (const int*)  d_in[0];
    const int*   ei       = (const int*)  d_in[1];   // [2,E]
    const int*   attr     = (const int*)  d_in[2];   // [E,3]
    const int*   batch    = (const int*)  d_in[3];
    const float* atom_emb = (const float*)d_in[4];   // [9,128,256]
    const float* bond_emb = (const float*)d_in[5];   // [L,3,16,3]
    const float* W        = (const float*)d_in[6];   // [L,256,256]
    const float* att_src  = (const float*)d_in[7];   // [L,256]
    const float* att_dst  = (const float*)d_in[8];
    const float* We       = (const float*)d_in[9];   // [L,3,256]
    const float* att_edge = (const float*)d_in[10];
    const float* bias     = (const float*)d_in[11];  // [L,256]
    const float* W1       = (const float*)d_in[12];  // [256,256]
    const float* b1       = (const float*)d_in[13];
    const float* W2       = (const float*)d_in[14];
    const float* b2       = (const float*)d_in[15];
    float* out = (float*)d_out;

    // -------- workspace carve-up (256B aligned) --------
    char* ws = (char*)d_ws;
    size_t off = 0;
    auto alloc = [&](size_t bytes) -> void* {
        void* p = ws + off;
        off = (off + bytes + 255) & ~(size_t)255;
        return p;
    };
    float*    h      = (float*)alloc((size_t)NN * EMB * 4);
    float*    xl     = (float*)alloc((size_t)NN * EMB * 4);
    float*    outb   = (float*)alloc((size_t)NN * EMB * 4);
    float*    alpha  = (float*)alloc((size_t)(EE + NN) * 4);
    float*    escore = (float*)alloc((size_t)EE * 4);
    float*    deg    = (float*)alloc((size_t)NN * 4);
    float*    sums   = (float*)alloc((size_t)NN * 4);
    float*    asrc   = (float*)alloc((size_t)NN * 4);
    float*    adst   = (float*)alloc((size_t)NN * 4);
    float*    denom  = (float*)alloc((size_t)NN * 4);
    unsigned* mkey   = (unsigned*)alloc((size_t)NN * 4);
    float*    lut    = (float*)alloc(64 * 4);
    float*    hg     = (float*)alloc((size_t)GG * EMB * 4);
    float*    cnt    = (float*)alloc((size_t)GG * 4);

    const int gNE   = NN;                       // N*EMB / 256
    const int gEpN  = (EE + NN + 255) / 256;
    const int gE    = (EE + 255) / 256;

    // Atom encoder
    k_atom<<<NN, 256, 0, stream>>>(x, atom_emb, h);

    for (int l = 0; l < LL; ++l) {
        const float* Wl    = W        + (size_t)l * EMB * EMB;
        const float* asl   = att_src  + (size_t)l * EMB;
        const float* adl   = att_dst  + (size_t)l * EMB;
        const float* Wel   = We       + (size_t)l * 3 * EMB;
        const float* ael   = att_edge + (size_t)l * EMB;
        const float* bl    = bias     + (size_t)l * EMB;
        const float* bondl = bond_emb + (size_t)l * 3 * BVOCAB * 3;

        k_init_layer<<<gNE, 256, 0, stream>>>(outb, deg, sums, denom, mkey);
        k_wa<<<1, 64, 0, stream>>>(Wel, ael, bondl, lut);
        k_edge_score<<<gE, 256, 0, stream>>>(ei, attr, lut, escore, deg, sums);
        // xl = h @ W[l]  (WMMA, 8 waves/block, A-tile reuse)
        k_gemm_wmma<false><<<dim3(NN / 16, EMB / 128), 256, 0, stream>>>(h, Wl, nullptr, xl);
        k_dots<<<(NN + 7) / 8, 256, 0, stream>>>(xl, asl, adl, asrc, adst);
        k_alpha_max<<<gEpN, 256, 0, stream>>>(ei, escore, deg, sums, asrc, adst, alpha, mkey);
        k_exp_sum<<<gEpN, 256, 0, stream>>>(ei, alpha, mkey, denom);
        k_scatter<<<EE + NN, 64, 0, stream>>>(ei, alpha, denom, xl, outb);
        k_post<<<gNE, 256, 0, stream>>>(outb, bl, h, (l < LL - 1) ? 1 : 0);
    }

    // Global mean pool + head MLP
    k_pool_init<<<GG, 256, 0, stream>>>(hg, cnt);
    k_pool_cnt<<<(NN + 255) / 256, 256, 0, stream>>>(batch, cnt);
    k_pool_sum<<<gNE, 256, 0, stream>>>(batch, h, hg);
    k_pool_fin<<<GG, 256, 0, stream>>>(hg, cnt);
    // t = hg @ W1 + b1  (WMMA, M=512) into xl scratch
    k_gemm_wmma<true><<<dim3(GG / 16, EMB / 128), 256, 0, stream>>>(hg, W1, b1, xl);
    k_head<<<GG, 256, 0, stream>>>(xl, W2, b2, out);
}
